// NemotronRouter_43946105372958
// MI455X (gfx1250) — matrix operand order repaired and checked by reading, not compiled, not run on hardware
//
#include <hip/hip_runtime.h>
#include <hip/hip_bf16.h>

typedef __attribute__((ext_vector_type(16))) __bf16 v16bf;
typedef __attribute__((ext_vector_type(4)))  __bf16 v4bf;
typedef __attribute__((ext_vector_type(8)))  float  v8f;

namespace {
constexpr int   kT       = 16384;   // tokens
constexpr int   kD       = 4096;    // d_model
constexpr int   kE       = 256;     // experts
constexpr int   kNG      = 8;       // groups
constexpr int   kEPG     = 32;      // experts per group
constexpr int   kTopkG   = 4;       // top groups
constexpr int   kTopK    = 8;       // top experts
constexpr float kScale   = 2.5f;
constexpr int   kTokWG   = 32;      // tokens per workgroup
constexpr int   kNTiles  = 4;       // 16-expert WMMA tiles per wave
}

// ---------------------------------------------------------------------------
// Pass 1: convert router weight fp32 -> bf16 once (2 MB in d_ws).
// ---------------------------------------------------------------------------
__global__ void nem_router_cvt_weight(const float* __restrict__ w,
                                      __bf16* __restrict__ wbf) {
  int i = (blockIdx.x * blockDim.x + threadIdx.x) * 4;
  float4 f = *(const float4*)(w + i);
  v4bf b;
  b[0] = (__bf16)f.x; b[1] = (__bf16)f.y;
  b[2] = (__bf16)f.z; b[3] = (__bf16)f.w;
  *(v4bf*)(wbf + i) = b;
}

// Load + convert one 16x32 bf16 A fragment for this lane.
__device__ __forceinline__ v16bf load_a_frag(const float* __restrict__ ap) {
  float4 f0 = *(const float4*)(ap);
  float4 f1 = *(const float4*)(ap + 4);
  float4 f2 = *(const float4*)(ap + 8);
  float4 f3 = *(const float4*)(ap + 12);
  v16bf a;
  a[0]  = (__bf16)f0.x; a[1]  = (__bf16)f0.y; a[2]  = (__bf16)f0.z; a[3]  = (__bf16)f0.w;
  a[4]  = (__bf16)f1.x; a[5]  = (__bf16)f1.y; a[6]  = (__bf16)f1.z; a[7]  = (__bf16)f1.w;
  a[8]  = (__bf16)f2.x; a[9]  = (__bf16)f2.y; a[10] = (__bf16)f2.z; a[11] = (__bf16)f2.w;
  a[12] = (__bf16)f3.x; a[13] = (__bf16)f3.y; a[14] = (__bf16)f3.z; a[15] = (__bf16)f3.w;
  return a;
}

__device__ __forceinline__ void load_b_set(const __bf16* const B[kNTiles],
                                           int k, v16bf out[kNTiles]) {
#pragma unroll
  for (int t = 0; t < kNTiles; ++t)
    out[t] = *(const v16bf*)(B[t] + k);   // two b128 loads, 32B aligned
}

// ---------------------------------------------------------------------------
// Pass 2: fused router GEMM (bf16 WMMA) + sigmoid + group-limited top-k.
// Workgroup: 256 threads = 8 waves, 32 tokens x 256 experts.
// Wave (mwave 0..1, nwave 0..3): 16 tokens x 64 experts (4 WMMA N-tiles).
// K-loop is unrolled x2 with ping-pong fragment sets (no rotation copies):
// each stage prefetches the other set's next fragments before its 4 WMMAs.
// ---------------------------------------------------------------------------
__global__ __launch_bounds__(256)
void nem_router_main(const float* __restrict__ hidden,
                     const __bf16* __restrict__ wbf,
                     const float* __restrict__ rbias,
                     const float* __restrict__ sbias,
                     int* __restrict__ idx_out,
                     float* __restrict__ w_out) {
  __shared__ float sc[kTokWG][kE + 1];   // +1 pad: conflict-free row reads

  const int tid   = threadIdx.x;
  const int wave  = tid >> 5;
  const int lane  = tid & 31;
  const int lrow  = lane & 15;
  const int lhi   = lane >> 4;          // 0: lanes 0-15, 1: lanes 16-31
  const int mwave = wave >> 2;          // 0..1
  const int nwave = wave & 3;           // 0..3
  const int tok0  = blockIdx.x * kTokWG;
  const int tokw  = tok0 + mwave * 16;
  const int e0    = nwave * 64;

  v8f acc[kNTiles] = {{}, {}, {}, {}};

  // A: hidden row for this lane (M = lrow); contiguous A k order used here (same as B):
  //   lanes 0-15 hold K k0..k0+15, lanes 16-31 hold k0+16..k0+31.
  const float* Arow = hidden + (size_t)(tokw + lrow) * kD + lhi * 16;
  // B: expert column = lane%16; lanes 0-15 hold K k0..k0+15, lanes 16-31 +16.
  // weight is [E][D] row-major, so each lane's 16 K values are contiguous.
  const __bf16* B[kNTiles];
#pragma unroll
  for (int t = 0; t < kNTiles; ++t)
    B[t] = wbf + (size_t)(e0 + 16 * t + lrow) * kD + lhi * 16;

  // Prologue: set A holds fragments for k = 0.
  v16bf aA = load_a_frag(Arow);
  v16bf bA[kNTiles];
  load_b_set(B, 0, bA);

  for (int k0 = 0; k0 < kD; k0 += 64) {
    // Stage 1: prefetch k0+32 into set B, compute on set A (k0).
    v16bf aB = load_a_frag(Arow + k0 + 32);
    v16bf bB[kNTiles];
    load_b_set(B, k0 + 32, bB);
#pragma unroll
    for (int t = 0; t < kNTiles; ++t)
      acc[t] = __builtin_amdgcn_wmma_f32_16x16x32_bf16(false, aA, false, bA[t],
                                                       (short)0, acc[t],
                                                       false, false);

    // Stage 2: prefetch k0+64 into set A (clamped dead load on last iter),
    // compute on set B (k0+32).
    const int kn = (k0 + 64 < kD) ? (k0 + 64) : 0;
    aA = load_a_frag(Arow + kn);
    load_b_set(B, kn, bA);
#pragma unroll
    for (int t = 0; t < kNTiles; ++t)
      acc[t] = __builtin_amdgcn_wmma_f32_16x16x32_bf16(false, aB, false, bB[t],
                                                       (short)0, acc[t],
                                                       false, false);
  }

  // C layout: VGPR r -> token row (lhi*8 + r), column N = lrow.
  {
    const int trow = mwave * 16 + lhi * 8;
#pragma unroll
    for (int t = 0; t < kNTiles; ++t) {
      const int e = e0 + 16 * t + lrow;
      const float rb = rbias[e], sb = sbias[e];
#pragma unroll
      for (int r = 0; r < 8; ++r)
        sc[trow + r][e] = 1.f / (1.f + __expf(-(acc[t][r] + rb))) + sb;
    }
  }
  __syncthreads();

  // Routing epilogue: one token per lane of wave 0.
  if (tid < kTokWG) {
    float* row = sc[tid];

    // Per-group sum of top-2 scores.
    float gw[kNG];
#pragma unroll
    for (int g = 0; g < kNG; ++g) {
      float m1 = -1e30f, m2 = -1e30f;
      for (int j = 0; j < kEPG; ++j) {
        float v = row[g * kEPG + j];
        float n1 = v > m1 ? v : m1;
        float n2 = v > m1 ? m1 : (v > m2 ? v : m2);
        m1 = n1; m2 = n2;
      }
      gw[g] = m1 + m2;
    }

    // Top-4 groups (strict > keeps lowest index on ties, like jax top_k).
    unsigned gmask = 0;
#pragma unroll
    for (int k = 0; k < kTopkG; ++k) {
      float best = -1e30f; int bg = 0;
#pragma unroll
      for (int g = 0; g < kNG; ++g) {
        bool take = (((gmask >> g) & 1u) == 0u) && (gw[g] > best);
        best = take ? gw[g] : best;
        bg   = take ? g : bg;
      }
      gmask |= (1u << bg);
    }

    // Top-8 experts over selected groups: single pass, sorted insertion.
    // Scores are sigmoid outputs (>0); unselected groups masked to -2.
    float tv[kTopK];
    int   ti[kTopK];
#pragma unroll
    for (int k = 0; k < kTopK; ++k) { tv[k] = -1.f; ti[k] = 0; }

    for (int j = 0; j < kE; ++j) {
      float v = ((gmask >> (j >> 5)) & 1u) ? row[j] : -2.f;
      if (v > tv[kTopK - 1]) {
#pragma unroll
        for (int k = kTopK - 1; k >= 1; --k) {
          bool gt  = v > tv[k];
          bool gtp = v > tv[k - 1];
          tv[k] = gt ? (gtp ? tv[k - 1] : v) : tv[k];
          ti[k] = gt ? (gtp ? ti[k - 1] : j) : ti[k];
        }
        if (v > tv[0]) { tv[0] = v; ti[0] = j; }
      }
    }

    float wsum = 0.f;
#pragma unroll
    for (int k = 0; k < kTopK; ++k) wsum += tv[k];
    const float inv = kScale / (wsum + 1e-20f);

    const int gtok = tok0 + tid;
#pragma unroll
    for (int k = 0; k < kTopK; ++k) {
      idx_out[gtok * kTopK + k] = ti[k];
      w_out[gtok * kTopK + k]   = tv[k] * inv;
    }
  }
}

// ---------------------------------------------------------------------------
extern "C" void kernel_launch(void* const* d_in, const int* in_sizes, int n_in,
                              void* d_out, int out_size, void* d_ws, size_t ws_size,
                              hipStream_t stream) {
  const float* hidden = (const float*)d_in[0];
  const float* weight = (const float*)d_in[1];
  const float* rbias  = (const float*)d_in[2];
  const float* sbias  = (const float*)d_in[3];

  __bf16* wbf = (__bf16*)d_ws;                       // 2 MB bf16 weight
  int*    idx_out = (int*)d_out;                     // [T, 8] int32 (first)
  float*  w_out   = (float*)d_out + (size_t)kT * kTopK;  // [T, 8] f32 (second)

  nem_router_cvt_weight<<<(kE * kD) / (256 * 4), 256, 0, stream>>>(weight, wbf);
  nem_router_main<<<kT / kTokWG, 256, 0, stream>>>(hidden, wbf, rbias, sbias,
                                                   idx_out, w_out);
  (void)in_sizes; (void)n_in; (void)out_size; (void)ws_size;
}
